// SimpleGRU_10093173145700
// MI455X (gfx1250) — matrix operand order repaired and checked by reading, not compiled, not run on hardware
//
#include <hip/hip_runtime.h>
#include <stdint.h>

#define HID   1024
#define IND   256
#define OUTD  256
#define BAT   64
#define SEQL  512
#define G3H   (3 * HID)
#define NBLK  64            // recurrence grid: one block per 16-wide hidden tile
#define LROW  1032          // padded LDS row length (bf16 elems): 2064B stride, conflict-free
#define LDSW_BYTES (3 * 16 * LROW * 2)

typedef __bf16 v16bf __attribute__((ext_vector_type(16)));
typedef float  v8f   __attribute__((ext_vector_type(8)));

union ABFrag { v16bf v; uint4 q[2]; };

static __device__ __forceinline__ unsigned short f2bf(float f) {
    unsigned int u = __float_as_uint(f);
    u += 0x7FFFu + ((u >> 16) & 1u);      // round-to-nearest-even
    return (unsigned short)(u >> 16);
}

static __device__ __forceinline__ float sigm(float x) {
    return 1.0f / (1.0f + __expf(-x));
}
static __device__ __forceinline__ float tanh_f(float x) {
    x = fminf(fmaxf(x, -15.0f), 15.0f);
    float e = __expf(2.0f * x);
    return (e - 1.0f) / (e + 1.0f);
}

// A fragment: 16x32 bf16 tile at (row0, k0) of row-major [*, ld].
// lanes 0-15 row M=lane, K=0..15; lanes 16-31 row M=lane-16, K=16..31 (matches load_b).
static __device__ __forceinline__ v16bf load_a(const unsigned short* __restrict__ base,
                                               int ld, int row0, int k0, int lane) {
    ABFrag f;
    const unsigned short* p = base + (size_t)(row0 + (lane & 15)) * ld
                                   + k0 + ((lane < 16) ? 0 : 16);
    f.q[0] = *(const uint4*)p;
    f.q[1] = *(const uint4*)(p + 8);
    return f.v;
}

// B fragment: 32x16 with B[k,n] = W[col0+n, k]; column n == contiguous row of row-major W.
// Lanes 0-15: K=0..15 of column (lane); lanes 16-31: K=16..31.
static __device__ __forceinline__ v16bf load_b(const unsigned short* __restrict__ w,
                                               int ld, int col0, int k0, int lane) {
    ABFrag f;
    const unsigned short* p = w + (size_t)(col0 + (lane & 15)) * ld
                                + k0 + ((lane < 16) ? 0 : 16);
    f.q[0] = *(const uint4*)p;
    f.q[1] = *(const uint4*)(p + 8);
    return f.v;
}

#define WMMA_BF16(a, b, c) \
    __builtin_amdgcn_wmma_f32_16x16x32_bf16(false, (a), false, (b), (short)0, (c), false, false)

// ---------------- elementwise conversion f32 -> bf16 ----------------
__global__ void k_cvt(const float* __restrict__ src, unsigned short* __restrict__ dst, int n) {
    int i = blockIdx.x * 256 + threadIdx.x;
    if (i < n) dst[i] = f2bf(src[i]);
}

// ---------------- h0 = 0, barrier counter = 0 ----------------
__global__ void k_init(float* __restrict__ hf, unsigned short* __restrict__ hb,
                       unsigned int* __restrict__ counter) {
    int i = blockIdx.x * 256 + threadIdx.x;
    if (i < BAT * HID) { hf[i] = 0.0f; hb[i] = 0; }
    if (i == 0) *counter = 0u;
}

// ---------------- gi = x * W_ih^T + b_ih, stored [S, B, 3H] ----------------
__global__ void k_gi(const unsigned short* __restrict__ xb,    // [B*S, 256] bf16
                     const unsigned short* __restrict__ wih,   // [3072, 256] bf16
                     const float* __restrict__ bih,
                     float* __restrict__ gi) {                 // [S, B, 3072] f32
    const int lane = threadIdx.x;
    const int m0 = blockIdx.y * 16;   // flattened row b*S + t
    const int g0 = blockIdx.x * 64;
    v8f acc[4] = {};
#pragma unroll
    for (int k = 0; k < IND; k += 32) {
        v16bf a = load_a(xb, IND, m0, k, lane);
#pragma unroll
        for (int s = 0; s < 4; ++s) {
            v16bf b = load_b(wih, IND, g0 + s * 16, k, lane);
            acc[s] = WMMA_BF16(a, b, acc[s]);
        }
    }
    const int n = lane & 15, mb = (lane >> 4) * 8;
#pragma unroll
    for (int s = 0; s < 4; ++s) {
        int col = g0 + s * 16 + n;
        float bias = bih[col];
#pragma unroll
        for (int v = 0; v < 8; ++v) {
            int m = m0 + mb + v;           // = b*SEQL + t (16 rows share one b)
            int b = m / SEQL, t = m % SEQL;
            gi[((size_t)t * BAT + b) * G3H + col] = acc[s][v] + bias;
        }
    }
}

// ---------------- persistent GRU recurrence ----------------
// 64 blocks (j-tiles) x 4 waves (batch tiles). W_hh slice async-staged into LDS once,
// reused for all 512 steps by all 4 waves. Grid-wide barrier per step.
__global__ void __launch_bounds__(128) k_gru(
        const float* __restrict__ gi,            // [S, B, 3H]
        const unsigned short* __restrict__ whh,  // [3072, 1024] bf16
        const float* __restrict__ bhh,
        float* __restrict__ hf0, unsigned short* __restrict__ hb0,
        float* __restrict__ hf1, unsigned short* __restrict__ hb1,
        unsigned int* __restrict__ counter) {
    extern __shared__ unsigned short ldsw[];     // [3][16][LROW] bf16
    const int tid  = threadIdx.x;
    const int lane = tid & 31;
    const int wv   = tid >> 5;
    const int j0   = blockIdx.x * 16;            // hidden tile (block)
    const int m0   = wv * 16;                    // batch tile (wave)

    // ---- stage W_hh[{r,z,n} rows j0..j0+15, all K] into LDS (async DMA path) ----
    {
        const unsigned ldsbase = __builtin_amdgcn_groupstaticsize();
        const int CHUNKS = 3 * 16 * (HID / 8);   // 16-byte chunks, 6144 total
        for (int i = tid; i < CHUNKS; i += 128) {
            int g   = i / (16 * (HID / 8));
            int rem = i % (16 * (HID / 8));
            int row = rem / (HID / 8);
            int ch  = rem % (HID / 8);
            unsigned ldsoff = ldsbase +
                (unsigned)((((g * 16 + row) * LROW) + ch * 8) * 2);
            unsigned long long ga = (unsigned long long)
                (whh + ((size_t)(g * HID + j0 + row)) * HID + (size_t)ch * 8);
            asm volatile("global_load_async_to_lds_b128 %0, %1, off"
                         :: "v"(ldsoff), "v"(ga) : "memory");
        }
        asm volatile("s_wait_asynccnt 0x0" ::: "memory");
        __syncthreads();
    }

    const int n = lane & 15, mb = (lane >> 4) * 8;
    const int j = j0 + n;
    const float bhr = bhh[j], bhz = bhh[HID + j], bhn = bhh[2 * HID + j];

    // LDS B-fragment addressing: column c = lane&15, K-half by lane group
    const int c_ = lane & 15;
    const int khalf = (lane < 16) ? 0 : 16;

    float* hin_f = hf0;  unsigned short* hin_b = hb0;
    float* hout_f = hf1; unsigned short* hout_b = hb1;

    for (int t = 0; t < SEQL; ++t) {
        v8f ar = {}, az = {}, an = {};
#pragma unroll 4
        for (int k = 0; k < HID; k += 32) {
            v16bf a = load_a(hin_b, HID, m0, k, lane);
            ABFrag br, bz, bn;
            {
                const int kb = k + khalf;
                const unsigned short* p0 = &ldsw[(0 * 16 + c_) * LROW + kb];
                const unsigned short* p1 = &ldsw[(1 * 16 + c_) * LROW + kb];
                const unsigned short* p2 = &ldsw[(2 * 16 + c_) * LROW + kb];
                br.q[0] = *(const uint4*)p0;  br.q[1] = *(const uint4*)(p0 + 8);
                bz.q[0] = *(const uint4*)p1;  bz.q[1] = *(const uint4*)(p1 + 8);
                bn.q[0] = *(const uint4*)p2;  bn.q[1] = *(const uint4*)(p2 + 8);
            }
            ar = WMMA_BF16(a, br.v, ar);
            az = WMMA_BF16(a, bz.v, az);
            an = WMMA_BF16(a, bn.v, an);
        }
#pragma unroll
        for (int v = 0; v < 8; ++v) {
            int b = m0 + mb + v;
            size_t g = ((size_t)t * BAT + b) * G3H;
            float r  = sigm(gi[g + j] + ar[v] + bhr);
            float z  = sigm(gi[g + HID + j] + az[v] + bhz);
            float nn = tanh_f(gi[g + 2 * HID + j] + r * (an[v] + bhn));
            float hp = hin_f[(size_t)b * HID + j];
            float hv = (1.0f - z) * nn + z * hp;
            hout_f[(size_t)b * HID + j] = hv;
            hout_b[(size_t)b * HID + j] = f2bf(hv);
        }
        // swap buffers
        { float* tf = hin_f; hin_f = hout_f; hout_f = tf;
          unsigned short* tb = hin_b; hin_b = hout_b; hout_b = tb; }
        // grid-wide barrier (monotonic counter, 64 co-resident blocks)
        __threadfence();
        __syncthreads();
        if (tid == 0) atomicAdd(counter, 1u);
        unsigned int target = (unsigned int)(t + 1) * NBLK;
        while (__hip_atomic_load(counter, __ATOMIC_ACQUIRE, __HIP_MEMORY_SCOPE_AGENT) < target) {
            __builtin_amdgcn_s_sleep(1);
        }
        __syncthreads();
    }
}

// ---------------- out = h_last * W_f^T + b_f ----------------
__global__ void k_final(const unsigned short* __restrict__ hb,  // [64, 1024] bf16
                        const unsigned short* __restrict__ wf,  // [256, 1024] bf16
                        const float* __restrict__ bf,
                        float* __restrict__ out) {              // [64, 256]
    const int lane = threadIdx.x;
    const int m0 = blockIdx.y * 16, o0 = blockIdx.x * 16;
    v8f acc = {};
#pragma unroll 4
    for (int k = 0; k < HID; k += 32) {
        v16bf a = load_a(hb, HID, m0, k, lane);
        v16bf b = load_b(wf, HID, o0, k, lane);
        acc = WMMA_BF16(a, b, acc);
    }
    const int n = lane & 15, mb = (lane >> 4) * 8;
    int o = o0 + n;
    float bias = bf[o];
#pragma unroll
    for (int v = 0; v < 8; ++v) {
        int m = m0 + mb + v;
        out[(size_t)m * OUTD + o] = acc[v] + bias;
    }
}

extern "C" void kernel_launch(void* const* d_in, const int* in_sizes, int n_in,
                              void* d_out, int out_size, void* d_ws, size_t ws_size,
                              hipStream_t stream) {
    (void)in_sizes; (void)n_in; (void)out_size; (void)ws_size;
    const float* x   = (const float*)d_in[0];
    const float* Wih = (const float*)d_in[1];
    const float* Whh = (const float*)d_in[2];
    const float* bih = (const float*)d_in[3];
    const float* bhh = (const float*)d_in[4];
    const float* Wf  = (const float*)d_in[5];
    const float* bf  = (const float*)d_in[6];
    float* out = (float*)d_out;

    char* ws = (char*)d_ws;
    size_t off = 0;
    auto take = [&](size_t bytes) -> char* {
        char* p = ws + off; off += (bytes + 255) & ~(size_t)255; return p;
    };
    float*          gi   = (float*)take((size_t)SEQL * BAT * G3H * 4);   // 402.7 MB
    unsigned short* xb   = (unsigned short*)take((size_t)BAT * SEQL * IND * 2);
    unsigned short* wihb = (unsigned short*)take((size_t)G3H * IND * 2);
    unsigned short* whhb = (unsigned short*)take((size_t)G3H * HID * 2);
    unsigned short* wfb  = (unsigned short*)take((size_t)OUTD * HID * 2);
    float*          hf0  = (float*)take((size_t)BAT * HID * 4);
    float*          hf1  = (float*)take((size_t)BAT * HID * 4);
    unsigned short* hb0  = (unsigned short*)take((size_t)BAT * HID * 2);
    unsigned short* hb1  = (unsigned short*)take((size_t)BAT * HID * 2);
    unsigned int*   ctr  = (unsigned int*)take(256);

    const int nx = BAT * SEQL * IND;
    k_cvt<<<(nx + 255) / 256, 256, 0, stream>>>(x, xb, nx);
    k_cvt<<<(G3H * IND + 255) / 256, 256, 0, stream>>>(Wih, wihb, G3H * IND);
    k_cvt<<<(G3H * HID + 255) / 256, 256, 0, stream>>>(Whh, whhb, G3H * HID);
    k_cvt<<<(OUTD * HID + 255) / 256, 256, 0, stream>>>(Wf, wfb, OUTD * HID);
    k_init<<<(BAT * HID + 255) / 256, 256, 0, stream>>>(hf0, hb0, ctr);

    k_gi<<<dim3(G3H / 64, (BAT * SEQL) / 16), 32, 0, stream>>>(xb, wihb, bih, gi);

    // after 512 steps (even), final h lands in buffer 0
    k_gru<<<dim3(NBLK), 128, LDSW_BYTES, stream>>>(gi, whhb, bhh,
                                                   hf0, hb0, hf1, hb1, ctr);

    k_final<<<dim3(OUTD / 16, BAT / 16), 32, 0, stream>>>(hb0, wfb, bf, out);
}
